// CausalSelfAttention_91225105367412
// MI455X (gfx1250) — matrix operand order repaired and checked by reading, not compiled, not run on hardware
//
#include <hip/hip_runtime.h>

// ---------------------------------------------------------------------------
// Causal self-attention on MI455X (gfx1250): bf16 WMMA + Tensor Data Mover.
//   B=2, S=2048, E=1024, H=16, D=64
// Stage 1: fp32 -> bf16 conversions (x copy; weights transposed to K-contig)
// Stage 2: QKV GEMM (TDM double-buffered tiles -> LDS, WMMA bf16, f32 acc)
// Stage 3: flash attention (streaming softmax, WMMA for QK^T and P*V)
// Stage 4: output projection GEMM -> fp32 d_out
// ---------------------------------------------------------------------------

typedef __bf16 bf16_t;
typedef __attribute__((ext_vector_type(16))) __bf16 v16bf;
typedef __attribute__((ext_vector_type(8)))  __bf16 v8bf;
typedef __attribute__((ext_vector_type(8)))  float  v8f;
typedef __attribute__((ext_vector_type(4)))  unsigned int v4u;
typedef __attribute__((ext_vector_type(8)))  int v8i;
typedef __attribute__((ext_vector_type(4)))  int v4i;

#define LOG2E 1.4426950408889634f

#if __has_builtin(__builtin_amdgcn_tensor_load_to_lds) && \
    __has_builtin(__builtin_amdgcn_s_wait_tensorcnt)
#define USE_TDM 1
#else
#define USE_TDM 0
#endif

static __device__ __forceinline__ bf16_t to_bf16(float f) {
  unsigned u = __builtin_bit_cast(unsigned, f);
  u += 0x7FFFu + ((u >> 16) & 1u);             // round-to-nearest-even
  unsigned short h = (unsigned short)(u >> 16);
  return __builtin_bit_cast(bf16_t, h);
}

// A fragment (16x32 bf16, M=rows): lane&15 = row; lane>>4 selects K 0..7/16..23
// vs 8..15/24..31.  Source must be row-major with K contiguous, ld in elems.
static __device__ __forceinline__ v16bf load_frag_a(const bf16_t* base, int ld, int lane) {
  const bf16_t* p = base + (size_t)(lane & 15) * ld + ((lane >> 4) * 8);
  v8bf lo = *(const v8bf*)(p);
  v8bf hi = *(const v8bf*)(p + 16);
  v16bf r;
#pragma unroll
  for (int i = 0; i < 8; ++i) { r[i] = lo[i]; r[i + 8] = hi[i]; }
  return r;
}

// B fragment (32x16 bf16, N=cols): lane&15 = column; lane>>4 selects K 0..7/16..23
// vs 8..15/24..31.  Source must be B^T (N-major, K contiguous), ld in elems.
static __device__ __forceinline__ v16bf load_frag_b(const bf16_t* base, int ld, int lane) {
  const bf16_t* p = base + (size_t)(lane & 15) * ld + ((lane >> 4) * 8);
  v8bf lo = *(const v8bf*)(p);
  v8bf hi = *(const v8bf*)(p + 16);
  v16bf r;
#pragma unroll
  for (int i = 0; i < 8; ++i) { r[i] = lo[i]; r[i + 8] = hi[i]; }
  return r;
}

// ------------------------------ conversions --------------------------------

__global__ __launch_bounds__(256) void k_f32_to_bf16(const float* __restrict__ in,
                                                     bf16_t* __restrict__ out, int n) {
  int i = blockIdx.x * 256 + threadIdx.x;
  if (i < n) out[i] = to_bf16(in[i]);
}

// in: [rows=K][cols=N] fp32   ->   out: [N][K] bf16   (out[n][k] = in[k][n])
__global__ __launch_bounds__(256) void k_transpose_to_bf16(const float* __restrict__ in,
                                                           bf16_t* __restrict__ out,
                                                           int rows, int cols) {
  int i = blockIdx.x * 256 + threadIdx.x;
  if (i < rows * cols) {
    int n = i / rows, k = i - n * rows;
    out[i] = to_bf16(in[(size_t)k * cols + n]);
  }
}

// ------------------------------ GEMM core ----------------------------------
// C[128x128] block, 8 waves (4 in M x 2 in N), wave tile 32x64.
// A: [M][K] bf16 row-major.  Bt: [N][K] bf16 (i.e. B transposed).
#define SLDA 40                  // LDS row stride in elems (80B, 16B aligned)
#define TILE_E (128 * SLDA)      // elems per LDS tile buffer
#define TILE_B (TILE_E * 2)      // bytes per LDS tile buffer

#if USE_TDM
// TDM descriptor: 2D tile, 128 rows x 32 bf16 elems, global row stride = ldg
// elems, LDS destination padded from 64B rows to 80B rows (pad 4 dwords after
// every 16 dwords) to match SLDA.  ISA D# layout per CDNA5 ISA ch.8.
static __device__ __forceinline__ void tdm_load_tile_128x32(const bf16_t* gsrc,
                                                            unsigned lds_addr,
                                                            unsigned ldg_elems) {
  unsigned long long ga = (unsigned long long)(uintptr_t)gsrc;
  v4u g0;
  g0[0] = 1u;                                            // count=1, user mode
  g0[1] = lds_addr;                                      // [63:32]  lds_addr
  g0[2] = (unsigned)(ga & 0xFFFFFFFFu);                  // [95:64]  gaddr lo
  g0[3] = (unsigned)((ga >> 32) & 0x1FFFFFFu) | (2u << 30); // gaddr hi | type=2
  v8i g1;
  g1[0] = (1 << 16)      // data_size = 2 bytes
        | (1 << 20)      // pad_enable
        | (3 << 22)      // pad_interval: 16 dwords (=64B row)
        | (3 << 25);     // pad_amount:   4 dwords (=16B pad -> 80B stride)
  g1[1] = (32 << 16);    // [47:32] atomic addr=0 | [63:48] tensor_dim0.lo = 32
  g1[2] = (128 << 16);   // [79:64] tensor_dim0.hi=0 | [95:80] tensor_dim1.lo=128
  g1[3] = (32 << 16);    // [111:96] tensor_dim1.hi=0 | [127:112] tile_dim0 = 32
  g1[4] = 128;           // [143:128] tile_dim1 = 128 | [159:144] tile_dim2 = 0
  g1[5] = (int)ldg_elems;// [191:160] tensor_dim0_stride lo
  g1[6] = 0;             // stride hi | tensor_dim1_stride lo
  g1[7] = 0;
  v4i gz4 = {};          // VADDR2/VADDR3 groups: NULL for <=2D tensors
  v8i gz8 = {};
  __builtin_amdgcn_tensor_load_to_lds(g0, g1, gz4, gz4, gz8, 0);
}
#endif

static __device__ __forceinline__ void gemm_block(const bf16_t* __restrict__ A,
                                                  const bf16_t* __restrict__ Bt,
                                                  int K, int blockM, int blockN,
                                                  bf16_t* sA, bf16_t* sB,
                                                  v8f acc[2][4]) {
  const int tid  = threadIdx.x;
  const int lane = tid & 31;
  const int wave = tid >> 5;
  const int wm   = wave >> 1;   // 0..3
  const int wn   = wave & 1;    // 0..1

  v8f z = {};
#pragma unroll
  for (int i = 0; i < 2; ++i)
#pragma unroll
    for (int j = 0; j < 4; ++j) acc[i][j] = z;

#if USE_TDM
  const unsigned ldsA = (unsigned)(uintptr_t)sA;
  const unsigned ldsB = (unsigned)(uintptr_t)sB;
  if (wave == 0) {   // tile 0 -> buffer 0 (TDM ignores EXEC; one issue per wave)
    tdm_load_tile_128x32(A  + (size_t)blockM * K, ldsA, (unsigned)K);
    tdm_load_tile_128x32(Bt + (size_t)blockN * K, ldsB, (unsigned)K);
  }
  for (int k0 = 0; k0 < K; k0 += 32) {
    const int cur = (k0 >> 5) & 1;
    if (wave == 0) __builtin_amdgcn_s_wait_tensorcnt(0);  // tile k0 resident
    __syncthreads();  // everyone: data ready; prev buffer's readers drained
    if (wave == 0 && (k0 + 32) < K) {
      const int nxt = cur ^ 1;
      tdm_load_tile_128x32(A  + (size_t)blockM * K + k0 + 32,
                           ldsA + (unsigned)(nxt * TILE_B), (unsigned)K);
      tdm_load_tile_128x32(Bt + (size_t)blockN * K + k0 + 32,
                           ldsB + (unsigned)(nxt * TILE_B), (unsigned)K);
    }
    const bf16_t* cA = sA + cur * TILE_E;
    const bf16_t* cB = sB + cur * TILE_E;
    v16bf af[2], bfr[4];
#pragma unroll
    for (int i = 0; i < 2; ++i) af[i]  = load_frag_a(&cA[(wm * 32 + i * 16) * SLDA], SLDA, lane);
#pragma unroll
    for (int j = 0; j < 4; ++j) bfr[j] = load_frag_b(&cB[(wn * 64 + j * 16) * SLDA], SLDA, lane);
#pragma unroll
    for (int i = 0; i < 2; ++i)
#pragma unroll
      for (int j = 0; j < 4; ++j)
        acc[i][j] = __builtin_amdgcn_wmma_f32_16x16x32_bf16(
            false, af[i], false, bfr[j], (short)0, acc[i][j], false, false);
  }
#else
  for (int k0 = 0; k0 < K; k0 += 32) {
#pragma unroll
    for (int c = 0; c < 2; ++c) {
      int id  = tid + c * 256;          // 0..511
      int r   = id >> 2;                // 0..127
      int col = (id & 3) * 8;           // 0,8,16,24
      *(v8bf*)&sA[r * SLDA + col] = *(const v8bf*)&A[(size_t)(blockM + r) * K + k0 + col];
      *(v8bf*)&sB[r * SLDA + col] = *(const v8bf*)&Bt[(size_t)(blockN + r) * K + k0 + col];
    }
    __syncthreads();
    v16bf af[2], bfr[4];
#pragma unroll
    for (int i = 0; i < 2; ++i) af[i]  = load_frag_a(&sA[(wm * 32 + i * 16) * SLDA], SLDA, lane);
#pragma unroll
    for (int j = 0; j < 4; ++j) bfr[j] = load_frag_b(&sB[(wn * 64 + j * 16) * SLDA], SLDA, lane);
#pragma unroll
    for (int i = 0; i < 2; ++i)
#pragma unroll
      for (int j = 0; j < 4; ++j)
        acc[i][j] = __builtin_amdgcn_wmma_f32_16x16x32_bf16(
            false, af[i], false, bfr[j], (short)0, acc[i][j], false, false);
    __syncthreads();
  }
#endif
}

// ------------------------------ QKV GEMM -----------------------------------
// X:[4096][1024]bf16  WqkvT:[3072][1024]bf16 -> Q/K/V [b=2][h=16][s=2048][d=64]
__global__ __launch_bounds__(256) void k_gemm_qkv(const bf16_t* __restrict__ X,
                                                  const bf16_t* __restrict__ Wt,
                                                  const float* __restrict__ bias,
                                                  bf16_t* __restrict__ Qo,
                                                  bf16_t* __restrict__ Ko,
                                                  bf16_t* __restrict__ Vo) {
  __shared__ bf16_t sA[2 * TILE_E];
  __shared__ bf16_t sB[2 * TILE_E];
  v8f acc[2][4];
  const int blockM = blockIdx.x * 128;
  const int blockN = blockIdx.y * 128;
  gemm_block(X, Wt, 1024, blockM, blockN, sA, sB, acc);

  const int lane = threadIdx.x & 31;
  const int wave = threadIdx.x >> 5;
  const int wm = wave >> 1, wn = wave & 1;

#pragma unroll
  for (int i = 0; i < 2; ++i)
#pragma unroll
    for (int j = 0; j < 4; ++j) {
      int col = blockN + wn * 64 + j * 16 + (lane & 15);
      int t = col >> 10;           // 0=Q 1=K 2=V (uniform per block)
      int c = col & 1023;
      int h = c >> 6, d = c & 63;
      bf16_t* dst = (t == 0) ? Qo : ((t == 1) ? Ko : Vo);   // branch-free select
      float scl = (t == 0) ? 0.125f : 1.0f;                 // 1/sqrt(64) on Q
      float bv = bias[col];
#pragma unroll
      for (int r = 0; r < 8; ++r) {
        int row = blockM + wm * 32 + i * 16 + ((lane >> 4) * 8) + r;
        int b = row >> 11, s = row & 2047;
        size_t off = ((size_t)((b * 16 + h) * 2048 + s)) * 64 + d;
        dst[off] = to_bf16((acc[i][j][r] + bv) * scl);
      }
    }
}

// ---------------------------- flash attention ------------------------------
// One block = 64 query rows of one (b,h); 4 waves x 16 rows each.
// Key blocks of 32; S=Q*K^T via WMMA (K loaded directly: storage IS B^T),
// streaming softmax, P staged through LDS into A-layout, O += P*V via WMMA.
__global__ __launch_bounds__(128) void k_attn(const bf16_t* __restrict__ Q,
                                              const bf16_t* __restrict__ K,
                                              const bf16_t* __restrict__ V,
                                              bf16_t* __restrict__ Ctx) {
  __shared__ bf16_t sVt[64 * SLDA];        // [d][key] transposed V block
  __shared__ bf16_t sP[4 * 16 * SLDA];     // per-wave P staging (16x32)

  const int lane = threadIdx.x & 31;
  const int wave = threadIdx.x >> 5;
  const int bh = blockIdx.y;               // b*16 + h
  const int h  = bh & 15;
  const int b  = bh >> 4;
  const int q0 = blockIdx.x * 64;
  const size_t base = (size_t)bh * 2048 * 64;
  const int qr = q0 + wave * 16;

  // Q fragments for this wave's 16 rows (d = 0..63 -> two K-steps of 32)
  v16bf qa[2];
  qa[0] = load_frag_a(Q + base + (size_t)qr * 64,      64, lane);
  qa[1] = load_frag_a(Q + base + (size_t)qr * 64 + 32, 64, lane);

  v8f o[4];
  v8f z = {};
#pragma unroll
  for (int t = 0; t < 4; ++t) o[t] = z;
  float mrow[8], lrow[8];
#pragma unroll
  for (int r = 0; r < 8; ++r) { mrow[r] = -__builtin_inff(); lrow[r] = 0.0f; }

  for (int j0 = 0; j0 < q0 + 64; j0 += 32) {
    // prefetch next K/V block into cache (global_prefetch_b8)
    if (j0 + 32 < q0 + 64) {
      const char* pk = (const char*)(K + base + (size_t)(j0 + 32) * 64);
      const char* pv = (const char*)(V + base + (size_t)(j0 + 32) * 64);
      __builtin_prefetch(pk + threadIdx.x * 32, 0, 3);
      __builtin_prefetch(pv + threadIdx.x * 32, 0, 3);
    }

    // load V block [32 keys][64 d] transposed into sVt[d][key]
#pragma unroll
    for (int c = 0; c < 2; ++c) {
      int id   = threadIdx.x + c * 128;    // 0..255
      int key  = id >> 3;                  // 0..31
      int dseg = (id & 7) * 8;
      v8bf vv = *(const v8bf*)&V[base + (size_t)(j0 + key) * 64 + dseg];
#pragma unroll
      for (int e = 0; e < 8; ++e) sVt[(dseg + e) * SLDA + key] = vv[e];
    }
    __syncthreads();

    // S = Q * K^T  (two 16-col key tiles, K-dim 64 = 2 wmma steps)
    v8f s[2];
#pragma unroll
    for (int jt = 0; jt < 2; ++jt) {
      v8f sa = z;
#pragma unroll
      for (int ks = 0; ks < 2; ++ks) {
        v16bf kb = load_frag_b(K + base + (size_t)(j0 + jt * 16) * 64 + ks * 32, 64, lane);
        sa = __builtin_amdgcn_wmma_f32_16x16x32_bf16(false, qa[ks], false, kb,
                                                     (short)0, sa, false, false);
      }
      s[jt] = sa;
    }

    // causal mask + online softmax (rows live in 16-lane half-groups)
    float p0v[8], p1v[8];
#pragma unroll
    for (int r = 0; r < 8; ++r) {
      int row = qr + ((lane >> 4) * 8) + r;
      int c0 = j0 + (lane & 15);
      float v0 = (c0      > row) ? -__builtin_inff() : s[0][r];
      float v1 = (c0 + 16 > row) ? -__builtin_inff() : s[1][r];
      float mx = fmaxf(v0, v1);
#pragma unroll
      for (int msk = 1; msk <= 8; msk <<= 1) mx = fmaxf(mx, __shfl_xor(mx, msk, 32));
      float mnew = fmaxf(mrow[r], mx);
      float sc = exp2f((mrow[r] - mnew) * LOG2E);
      float p0 = exp2f((v0 - mnew) * LOG2E);
      float p1 = exp2f((v1 - mnew) * LOG2E);
      float ps = p0 + p1;
#pragma unroll
      for (int msk = 1; msk <= 8; msk <<= 1) ps += __shfl_xor(ps, msk, 32);
      lrow[r] = lrow[r] * sc + ps;
      mrow[r] = mnew;
      p0v[r] = p0; p1v[r] = p1;
#pragma unroll
      for (int t = 0; t < 4; ++t) o[t][r] *= sc;
    }

    // stage P (C-layout) -> LDS -> reload as A fragment
    bf16_t* myP = &sP[wave * 16 * SLDA];
#pragma unroll
    for (int r = 0; r < 8; ++r) {
      int rr = ((lane >> 4) * 8) + r;
      myP[rr * SLDA +      (lane & 15)] = to_bf16(p0v[r]);
      myP[rr * SLDA + 16 + (lane & 15)] = to_bf16(p1v[r]);
    }
    __syncthreads();

    v16bf pa = load_frag_a(myP, SLDA, lane);
#pragma unroll
    for (int t = 0; t < 4; ++t) {
      v16bf vb = load_frag_b(&sVt[(t * 16) * SLDA], SLDA, lane);
      o[t] = __builtin_amdgcn_wmma_f32_16x16x32_bf16(false, pa, false, vb,
                                                     (short)0, o[t], false, false);
    }
    __syncthreads();   // before next iteration overwrites sVt/sP
  }

  // normalize and write context [b*2048+s][h*64+d] bf16
#pragma unroll
  for (int t = 0; t < 4; ++t) {
    int d = t * 16 + (lane & 15);
#pragma unroll
    for (int r = 0; r < 8; ++r) {
      int row = qr + ((lane >> 4) * 8) + r;
      float val = o[t][r] / lrow[r];
      Ctx[((size_t)(b * 2048 + row)) * 1024 + h * 64 + d] = to_bf16(val);
    }
  }
}

// ---------------------------- output projection ----------------------------
__global__ __launch_bounds__(256) void k_gemm_out(const bf16_t* __restrict__ Ctx,
                                                  const bf16_t* __restrict__ WoT,
                                                  const float* __restrict__ bias,
                                                  float* __restrict__ Out) {
  __shared__ bf16_t sA[2 * TILE_E];
  __shared__ bf16_t sB[2 * TILE_E];
  v8f acc[2][4];
  const int blockM = blockIdx.x * 128;
  const int blockN = blockIdx.y * 128;
  gemm_block(Ctx, WoT, 1024, blockM, blockN, sA, sB, acc);

  const int lane = threadIdx.x & 31;
  const int wave = threadIdx.x >> 5;
  const int wm = wave >> 1, wn = wave & 1;

#pragma unroll
  for (int i = 0; i < 2; ++i)
#pragma unroll
    for (int j = 0; j < 4; ++j) {
      int col = blockN + wn * 64 + j * 16 + (lane & 15);
      float bv = bias[col];
#pragma unroll
      for (int r = 0; r < 8; ++r) {
        int row = blockM + wm * 32 + i * 16 + ((lane >> 4) * 8) + r;
        Out[(size_t)row * 1024 + col] = acc[i][j][r] + bv;
      }
    }
}

// ------------------------------ launcher -----------------------------------
extern "C" void kernel_launch(void* const* d_in, const int* in_sizes, int n_in,
                              void* d_out, int out_size, void* d_ws, size_t ws_size,
                              hipStream_t stream) {
  (void)in_sizes; (void)n_in; (void)out_size; (void)ws_size;
  const float* x     = (const float*)d_in[0];   // [2,2048,1024]
  const float* w_qkv = (const float*)d_in[1];   // [1024,3072]
  const float* b_qkv = (const float*)d_in[2];   // [3072]
  const float* w_o   = (const float*)d_in[3];   // [1024,1024]
  const float* b_o   = (const float*)d_in[4];   // [1024]

  char* ws = (char*)d_ws;
  bf16_t* Xb  = (bf16_t*)(ws);                        //  8 MiB: 4096x1024
  bf16_t* WqT = (bf16_t*)(ws + ((size_t)8  << 20));   //  6 MiB: 3072x1024
  bf16_t* WoT = (bf16_t*)(ws + ((size_t)14 << 20));   //  2 MiB: 1024x1024
  bf16_t* Qb  = (bf16_t*)(ws + ((size_t)16 << 20));   //  8 MiB: [2,16,2048,64]
  bf16_t* Kb  = (bf16_t*)(ws + ((size_t)24 << 20));   //  8 MiB
  bf16_t* Vb  = (bf16_t*)(ws + ((size_t)32 << 20));   //  8 MiB
  bf16_t* Ctx = (bf16_t*)(ws + ((size_t)40 << 20));   //  8 MiB: 4096x1024

  k_f32_to_bf16<<<(4096 * 1024 + 255) / 256, 256, 0, stream>>>(x, Xb, 4096 * 1024);
  k_transpose_to_bf16<<<(1024 * 3072 + 255) / 256, 256, 0, stream>>>(w_qkv, WqT, 1024, 3072);
  k_transpose_to_bf16<<<(1024 * 1024 + 255) / 256, 256, 0, stream>>>(w_o, WoT, 1024, 1024);

  k_gemm_qkv<<<dim3(32, 24), 256, 0, stream>>>(Xb, WqT, b_qkv, Qb, Kb, Vb);
  k_attn    <<<dim3(32, 32), 128, 0, stream>>>(Qb, Kb, Vb, Ctx);
  k_gemm_out<<<dim3(32, 8),  256, 0, stream>>>(Ctx, WoT, b_o, (float*)d_out);
}
